// MixtralDecoderLayer_41686952575382
// MI455X (gfx1250) — compile-verified, matrix operand order checked by reading, not run on hardware
//
#include <hip/hip_runtime.h>
#include <hip/hip_bf16.h>
#include <math.h>

// ---------------- problem dims (fixed by reference) ----------------
static constexpr int Bb   = 2;
static constexpr int Ss   = 1024;
static constexpr int Hh   = 1024;
static constexpr int NHq  = 16;
static constexpr int NKVq = 4;
static constexpr int HDd  = 64;
static constexpr int Ii   = 3584;
static constexpr int Ee   = 8;
static constexpr int Tt   = Bb * Ss;                   // 2048 tokens
static constexpr int QKVD = (NHq + 2 * NKVq) * HDd;    // 1536

// ---------------- WMMA types ----------------
typedef __attribute__((ext_vector_type(16))) __bf16 v16bf;
typedef __attribute__((ext_vector_type(8)))  float  v8f;

union FragU { v16bf v; uint4 q[2]; };
union Pack4 { __bf16 h[4]; uint2 u; };

// =======================================================================
// Tiled bf16 WMMA GEMM with double-buffered LDS:  C[M,N] = A[M,K] * B
//   BT==0 : B stored [K,N] row-major (w_qkv, w_o)
//   BT==1 : B stored [N,K] row-major (w1/w3/w2 -- x @ W^T)
//   EPI==0: C = acc
//   EPI==1: C = acc + aux[row*N+col]           (residual add)
//   EPI==2: C[row,col] += aux[row*Ee] * acc    (router-weighted combine)
// Block tile 128x128x32, 256 threads = 8 waves (2 M x 4 N),
// wave tile 64x32 = 4x2 WMMA 16x16 tiles -> 8 v_wmma per K-step.
// Next tile's global loads are issued BEFORE the current tile's WMMAs so
// HBM latency overlaps the matrix pipe; loads are consumed (waited) only
// at the convert+store into the ping-pong buffer.
// =======================================================================
template <int BT, int EPI>
__global__ __launch_bounds__(256)
void gemm_bf16_wmma(const float* __restrict__ A, const float* __restrict__ B,
                    float* __restrict__ C, const float* __restrict__ aux,
                    int M, int N, int K) {
  constexpr int LDT = 40;  // padded LDS row stride in bf16 (80B: 16B-aligned, conflict-free)
  __shared__ __align__(16) __bf16 ldsA[2][128 * LDT];
  __shared__ __align__(16) __bf16 ldsB[2][128 * LDT];

  const int tid   = threadIdx.x;
  const int lane  = tid & 31;
  const int wave  = tid >> 5;
  const int wm    = wave & 1;      // wave row (64 M-rows each)
  const int wn    = wave >> 1;     // wave col (32 N-cols each)
  const int lrow  = lane & 15;
  const int khalf = lane >> 4;

  const int bM = blockIdx.y * 128;
  const int bN = blockIdx.x * 128;

  v8f acc[4][2];
#pragma unroll
  for (int i = 0; i < 4; ++i)
#pragma unroll
    for (int j = 0; j < 2; ++j)
#pragma unroll
      for (int r = 0; r < 8; ++r) acc[i][j][r] = 0.0f;

  float4 ra[4], rb[4];

  // issue global loads for the K-tile starting at kk into registers
  auto g_load = [&](int kk) {
#pragma unroll
    for (int c = 0; c < 4; ++c) {
      const int lin = (c * 256 + tid) * 4;       // 0..4095, step 4
      const int row = lin >> 5, col = lin & 31;  // A: row 0..127, k 0..31
      ra[c] = *(const float4*)(A + (size_t)(bM + row) * K + kk + col);
      if (BT) {
        rb[c] = *(const float4*)(B + (size_t)(bN + row) * K + kk + col);
      } else {
        const int krow = lin >> 7, ncol = lin & 127;
        rb[c] = *(const float4*)(B + (size_t)(kk + krow) * N + bN + ncol);
      }
    }
  };

  // convert to bf16 and store into LDS buffer `buf`
  auto s_store = [&](int buf) {
#pragma unroll
    for (int c = 0; c < 4; ++c) {
      const int lin = (c * 256 + tid) * 4;
      const int row = lin >> 5, col = lin & 31;
      Pack4 pa;
      pa.h[0] = (__bf16)ra[c].x; pa.h[1] = (__bf16)ra[c].y;
      pa.h[2] = (__bf16)ra[c].z; pa.h[3] = (__bf16)ra[c].w;
      *(uint2*)&ldsA[buf][row * LDT + col] = pa.u;
      if (BT) {  // K contiguous: packed 8B store
        Pack4 pb;
        pb.h[0] = (__bf16)rb[c].x; pb.h[1] = (__bf16)rb[c].y;
        pb.h[2] = (__bf16)rb[c].z; pb.h[3] = (__bf16)rb[c].w;
        *(uint2*)&ldsB[buf][row * LDT + col] = pb.u;
      } else {   // transpose into [n][k]: scattered b16 stores
        const int krow = lin >> 7, ncol = lin & 127;
        ldsB[buf][(ncol + 0) * LDT + krow] = (__bf16)rb[c].x;
        ldsB[buf][(ncol + 1) * LDT + krow] = (__bf16)rb[c].y;
        ldsB[buf][(ncol + 2) * LDT + krow] = (__bf16)rb[c].z;
        ldsB[buf][(ncol + 3) * LDT + krow] = (__bf16)rb[c].w;
      }
    }
  };

  // prologue: stage tile 0
  g_load(0);
  s_store(0);
  __syncthreads();

  const int nk = K >> 5;
  for (int t = 0; t < nk; ++t) {
    const int cur = t & 1;
    if (t + 1 < nk) {
      g_load((t + 1) << 5);   // loads in flight during the WMMAs below
      __builtin_prefetch(A + (size_t)(bM + (tid & 127)) * K + ((t + 2) << 5), 0, 1);
    }

    // ---- fragment gathers (ISA 16-bit A 16x32 layout) ----
    FragU a[4], b2[2];
#pragma unroll
    for (int i = 0; i < 4; ++i) {
      const __bf16* p = &ldsA[cur][(wm * 64 + i * 16 + lrow) * LDT + khalf * 8];
      a[i].q[0] = *(const uint4*)(p);        // K = khalf*8 .. +7
      a[i].q[1] = *(const uint4*)(p + 16);   // K = 16 + khalf*8 .. +7
    }
#pragma unroll
    for (int j = 0; j < 2; ++j) {
      const __bf16* p = &ldsB[cur][(wn * 32 + j * 16 + lrow) * LDT + khalf * 8];
      b2[j].q[0] = *(const uint4*)(p);
      b2[j].q[1] = *(const uint4*)(p + 16);
    }

#pragma unroll
    for (int i = 0; i < 4; ++i)
#pragma unroll
      for (int j = 0; j < 2; ++j)
        acc[i][j] = __builtin_amdgcn_wmma_f32_16x16x32_bf16(
            false, a[i].v, false, b2[j].v, (short)0, acc[i][j], false, false);

    if (t + 1 < nk) s_store(cur ^ 1);  // consumes ra/rb (waits loads here)
    __syncthreads();
  }

  // ---- epilogue: C/D layout -> row = r + 8*(lane/16), col = lane%16 ----
#pragma unroll
  for (int i = 0; i < 4; ++i)
#pragma unroll
    for (int j = 0; j < 2; ++j)
#pragma unroll
      for (int r = 0; r < 8; ++r) {
        const int row = bM + wm * 64 + i * 16 + r + (khalf << 3);
        const int col = bN + wn * 32 + j * 16 + lrow;
        const size_t idx = (size_t)row * N + col;
        float v = acc[i][j][r];
        if (EPI == 1) v += aux[idx];
        if (EPI == 2) C[idx] += aux[(size_t)row * Ee] * v;
        else          C[idx] = v;
      }
}

// =======================================================================
// RMSNorm: one block (256 thr) per token row of H=1024 (4 elems/thread)
// =======================================================================
__global__ __launch_bounds__(256)
void rmsnorm_kernel(const float* __restrict__ x, const float* __restrict__ w,
                    float* __restrict__ y) {
  const int row = blockIdx.x;
  const int tid = threadIdx.x;
  __shared__ float red[256];
  const float* xr = x + (size_t)row * Hh;
  const float4 xv = *(const float4*)(xr + tid * 4);
  red[tid] = xv.x * xv.x + xv.y * xv.y + xv.z * xv.z + xv.w * xv.w;
  __syncthreads();
  for (int o = 128; o > 0; o >>= 1) {
    if (tid < o) red[tid] += red[tid + o];
    __syncthreads();
  }
  const float rs = rsqrtf(red[0] * (1.0f / Hh) + 1e-5f);
  const float4 wv = *(const float4*)(w + tid * 4);
  float4 ov;
  ov.x = xv.x * rs * wv.x; ov.y = xv.y * rs * wv.y;
  ov.z = xv.z * rs * wv.z; ov.w = xv.w * rs * wv.w;
  *(float4*)(y + (size_t)row * Hh + tid * 4) = ov;
}

// =======================================================================
// RoPE in-place on qkv buffer: q heads 0..15, k heads 16..19; pair d,d+32
// =======================================================================
__global__ void rope_kernel(const int* __restrict__ positions,
                            float* __restrict__ qkv) {
  const int idx = blockIdx.x * blockDim.x + threadIdx.x;
  if (idx >= Tt * 20 * 32) return;
  const int d = idx & 31;
  const int h = (idx >> 5) % 20;
  const int t = idx / (20 * 32);
  const float inv = __expf(-logf(10000.0f) * (float)d * (1.0f / 32.0f));
  const float fr = (float)positions[t] * inv;
  const float c = __cosf(fr), s = __sinf(fr);
  const int col = (h < NHq) ? h * HDd : NHq * HDd + (h - NHq) * HDd;
  float* p = qkv + (size_t)t * QKVD + col;
  const float x1 = p[d], x2 = p[d + 32];
  p[d]      = x1 * c - x2 * s;
  p[d + 32] = x2 * c + x1 * s;
}

// =======================================================================
// Causal GQA attention, streaming softmax. Block = (4 q-heads x 64 q-rows)
// per (batch, kv-head, 64-row q tile). K/V tiles staged to LDS with the
// gfx1250 async-to-LDS path (GLOBAL_LOAD_ASYNC_TO_LDS_B128, ASYNCcnt),
// bypassing VGPRs entirely; s_wait_asynccnt + barrier publishes the tile.
// KV tile struct is the only LDS object -> K at LDS offset 0, V at +16384.
// =======================================================================
__global__ __launch_bounds__(256)
void attn_kernel(const float* __restrict__ qkv, float* __restrict__ ctxout) {
  const int bid = blockIdx.x;
  const int qt  = bid & 15;          // S/64 = 16
  const int kvh = (bid >> 4) & 3;
  const int b   = bid >> 6;
  const int tid = threadIdx.x;
  const int g   = tid >> 6;          // 0..3 q-head within group
  const int r   = tid & 63;
  const int qrow = qt * 64 + r;
  const int head = kvh * 4 + g;      // NH head index

  __shared__ __align__(16) float KVt[2][64][64];  // [0]=K tile, [1]=V tile

  const float* qp = qkv + (size_t)(b * Ss + qrow) * QKVD + head * HDd;
  float q[64], ctx[64];
#pragma unroll
  for (int d = 0; d < 64; ++d) { q[d] = qp[d] * 0.125f; ctx[d] = 0.0f; }
  float m = -1e30f, l = 0.0f;

  for (int kt = 0; kt <= qt; ++kt) {
    __syncthreads();  // previous tile fully consumed before overwrite
#pragma unroll
    for (int c = 0; c < 4; ++c) {
      const int lin  = (c * 256 + tid) * 4;
      const int trow = lin >> 6, dcol = lin & 63;
      const int tg   = kt * 64 + trow;
      const float* kp = qkv + (size_t)(b * Ss + tg) * QKVD +
                        NHq * HDd + kvh * HDd + dcol;
      const float* vp = kp + NKVq * HDd;
      const unsigned ldsK = (unsigned)((trow * 64 + dcol) * 4);  // KVt[0] base
      const unsigned ldsV = ldsK + 16384u;                       // KVt[1] base
      asm volatile("global_load_async_to_lds_b128 %0, %1, off"
                   :: "v"(ldsK), "v"(kp) : "memory");
      asm volatile("global_load_async_to_lds_b128 %0, %1, off"
                   :: "v"(ldsV), "v"(vp) : "memory");
    }
    asm volatile("s_wait_asynccnt 0x0" ::: "memory");
    __syncthreads();

    for (int c0 = 0; c0 < 64; c0 += 16) {
      float s[16];
#pragma unroll
      for (int u = 0; u < 16; ++u) {
        const int tg = kt * 64 + c0 + u;
        float acc = 0.0f;
#pragma unroll
        for (int d = 0; d < 64; ++d) acc += q[d] * KVt[0][c0 + u][d];
        s[u] = (tg <= qrow) ? acc : -1e30f;
      }
      float cm = m;
#pragma unroll
      for (int u = 0; u < 16; ++u) cm = fmaxf(cm, s[u]);
      const float corr = __expf(m - cm);
      m = cm; l *= corr;
#pragma unroll
      for (int d = 0; d < 64; ++d) ctx[d] *= corr;
#pragma unroll
      for (int u = 0; u < 16; ++u) {
        const float p = __expf(s[u] - cm);
        l += p;
#pragma unroll
        for (int d = 0; d < 64; ++d) ctx[d] += p * KVt[1][c0 + u][d];
      }
    }
  }
  const float inv = 1.0f / l;
  float* op = ctxout + (size_t)(b * Ss + qrow) * (NHq * HDd) + head * HDd;
#pragma unroll
  for (int d = 0; d < 64; ++d) op[d] = ctx[d] * inv;
}

// =======================================================================
// Router: per-token logits -> softmax -> top-2 -> renormalize -> wts[T,E]
// =======================================================================
__global__ void router_kernel(const float* __restrict__ h2,
                              const float* __restrict__ gate_w,
                              float* __restrict__ wts) {
  const int t = blockIdx.x * blockDim.x + threadIdx.x;
  if (t >= Tt) return;
  float lg[Ee];
#pragma unroll
  for (int e = 0; e < Ee; ++e) lg[e] = 0.0f;
  const float* x = h2 + (size_t)t * Hh;
  for (int h = 0; h < Hh; ++h) {
    const float xv = x[h];
    const float* gw = gate_w + (size_t)h * Ee;
#pragma unroll
    for (int e = 0; e < Ee; ++e) lg[e] += xv * gw[e];
  }
  float mx = lg[0];
#pragma unroll
  for (int e = 1; e < Ee; ++e) mx = fmaxf(mx, lg[e]);
  float pe[Ee];
#pragma unroll
  for (int e = 0; e < Ee; ++e) pe[e] = __expf(lg[e] - mx);
  int i1 = 0;
#pragma unroll
  for (int e = 1; e < Ee; ++e) if (pe[e] > pe[i1]) i1 = e;
  int i2 = (i1 == 0) ? 1 : 0;
#pragma unroll
  for (int e = 0; e < Ee; ++e) if (e != i1 && pe[e] > pe[i2]) i2 = e;
  const float inv = 1.0f / (pe[i1] + pe[i2]);
#pragma unroll
  for (int e = 0; e < Ee; ++e)
    wts[(size_t)t * Ee + e] = (e == i1) ? pe[i1] * inv
                            : (e == i2) ? pe[i2] * inv : 0.0f;
}

// =======================================================================
__global__ void silu_mul_kernel(float* __restrict__ G,
                                const float* __restrict__ U, size_t n) {
  const size_t i = (size_t)blockIdx.x * blockDim.x + threadIdx.x;
  if (i >= n) return;
  const float g = G[i];
  G[i] = (g / (1.0f + __expf(-g))) * U[i];
}

__global__ void zero_kernel(float* __restrict__ p, size_t n) {
  const size_t i = (size_t)blockIdx.x * blockDim.x + threadIdx.x;
  if (i < n) p[i] = 0.0f;
}

// =======================================================================
extern "C" void kernel_launch(void* const* d_in, const int* in_sizes, int n_in,
                              void* d_out, int out_size, void* d_ws, size_t ws_size,
                              hipStream_t stream) {
  const int*   positions = (const int*)d_in[0];
  const float* hidden    = (const float*)d_in[1];
  const float* w_qkv     = (const float*)d_in[2];
  const float* w_o       = (const float*)d_in[3];
  const float* gate_w    = (const float*)d_in[4];
  const float* w1        = (const float*)d_in[5];
  const float* w3        = (const float*)d_in[6];
  const float* w2        = (const float*)d_in[7];
  const float* ln1_w     = (const float*)d_in[8];
  const float* ln2_w     = (const float*)d_in[9];

  float* out      = (float*)d_out;                     // MoE output [T,H]
  float* residual = out + (size_t)Tt * Hh;             // second tuple element

  float* ws  = (float*)d_ws;
  float* h1  = ws;  ws += (size_t)Tt * Hh;
  float* qkv = ws;  ws += (size_t)Tt * QKVD;
  float* ctx = ws;  ws += (size_t)Tt * (NHq * HDd);
  float* h2  = ws;  ws += (size_t)Tt * Hh;
  float* wts = ws;  ws += (size_t)Tt * Ee;
  float* Gb  = ws;  ws += (size_t)Tt * Ii;
  float* Ub  = ws;  ws += (size_t)Tt * Ii;

  // 1) ln1
  rmsnorm_kernel<<<Tt, 256, 0, stream>>>(hidden, ln1_w, h1);
  // 2) qkv = h1 @ w_qkv  (B is [K=H, N=1536])
  gemm_bf16_wmma<0, 0><<<dim3(QKVD / 128, Tt / 128), 256, 0, stream>>>(
      h1, w_qkv, qkv, nullptr, Tt, QKVD, Hh);
  // 3) RoPE on q,k in place
  rope_kernel<<<(Tt * 20 * 32 + 255) / 256, 256, 0, stream>>>(positions, qkv);
  // 4) causal GQA attention (async-to-LDS K/V staging)
  attn_kernel<<<Bb * NKVq * (Ss / 64), 256, 0, stream>>>(qkv, ctx);
  // 5) residual = ctx @ w_o + hidden   (written into second half of d_out)
  gemm_bf16_wmma<0, 1><<<dim3(Hh / 128, Tt / 128), 256, 0, stream>>>(
      ctx, w_o, residual, hidden, Tt, Hh, NHq * HDd);
  // 6) ln2
  rmsnorm_kernel<<<Tt, 256, 0, stream>>>(residual, ln2_w, h2);
  // 7) router weights
  router_kernel<<<(Tt + 255) / 256, 256, 0, stream>>>(h2, gate_w, wts);
  // 8) zero MoE accumulator (harness poisons d_out)
  zero_kernel<<<((size_t)Tt * Hh + 255) / 256, 256, 0, stream>>>(out, (size_t)Tt * Hh);
  // 9) dense experts, router-weighted combine
  for (int e = 0; e < Ee; ++e) {
    const float* w1e = w1 + (size_t)e * Ii * Hh;   // [I,H] -> x @ w1^T : BT
    const float* w3e = w3 + (size_t)e * Ii * Hh;
    const float* w2e = w2 + (size_t)e * Hh * Ii;   // [H,I] -> gu @ w2^T : BT
    gemm_bf16_wmma<1, 0><<<dim3(Ii / 128, Tt / 128), 256, 0, stream>>>(
        h2, w1e, Gb, nullptr, Tt, Ii, Hh);
    gemm_bf16_wmma<1, 0><<<dim3(Ii / 128, Tt / 128), 256, 0, stream>>>(
        h2, w3e, Ub, nullptr, Tt, Ii, Hh);
    silu_mul_kernel<<<((size_t)Tt * Ii + 255) / 256, 256, 0, stream>>>(
        Gb, Ub, (size_t)Tt * Ii);
    gemm_bf16_wmma<1, 2><<<dim3(Hh / 128, Tt / 128), 256, 0, stream>>>(
        Gb, w2e, out, wts + e, Tt, Hh, Ii);
  }
}
